// NemotronFlashSDPAAttention_7292854468789
// MI455X (gfx1250) — matrix operand order repaired and checked by reading, not compiled, not run on hardware
//
#include <hip/hip_runtime.h>
#include <hip/hip_bf16.h>

typedef unsigned short u16;
typedef __attribute__((ext_vector_type(8)))  u16    v8u;
typedef __attribute__((ext_vector_type(16))) u16    v16u;
typedef __attribute__((ext_vector_type(16))) __bf16 v16bf;
typedef __attribute__((ext_vector_type(8)))  float  v8f;
typedef __attribute__((ext_vector_type(4)))  unsigned int u32x4;
typedef __attribute__((ext_vector_type(8)))  unsigned int u32x8;

#define S_LEN   2048
#define D_MODEL 4096
#define NH      32
#define NKVH    8
#define HDIM    128
#define EQ      (NH * HDIM)    // 4096
#define EKV     (NKVH * HDIM)  // 1024

// padded LDS tile geometry: 128 cols + 8 pad halves -> 136 halves/row
#define TROW    136
#define TILE_H  (128 * TROW)   // halves per 128-row tile

// ---------- helpers ----------

__device__ __forceinline__ u16 f2bf(float f) {
  union { float f; unsigned u; } x; x.f = f;
  unsigned u = x.u;
  u += 0x7fffu + ((u >> 16) & 1u);   // round-to-nearest-even
  return (u16)(u >> 16);
}

__device__ __forceinline__ v16bf mkfrag(v8u lo, v8u hi) {
  union { v16u u; v16bf h; } x;
#pragma unroll
  for (int i = 0; i < 8; ++i) { x.u[i] = lo[i]; x.u[i + 8] = hi[i]; }
  return x.h;
}

__device__ __forceinline__ v8f vzero8() {
  v8f z;
#pragma unroll
  for (int i = 0; i < 8; ++i) z[i] = 0.0f;
  return z;
}

__device__ __forceinline__ float rmax16(float v) {
#pragma unroll
  for (int m = 1; m <= 8; m <<= 1) v = fmaxf(v, __shfl_xor(v, m, 32));
  return v;
}
__device__ __forceinline__ float rsum16(float v) {
#pragma unroll
  for (int m = 1; m <= 8; m <<= 1) v += __shfl_xor(v, m, 32);
  return v;
}

// ---------- TDM: async 2D tile load Global -> LDS (CDNA5 Tensor Data Mover) ----
// 2D tile of 2-byte elements, tile_w x tile_h, row stride in elements.
// LDS padding: +4 DWORDs after every 64 DWORDs (256B row) -> 136-half row stride.
// Descriptor per CDNA5 ISA ch.8 (group0: count/lds/global/type, group1: dims).

__device__ __forceinline__ void tdm_load_2d_bf16(unsigned lds_off_bytes,
                                                 const u16* gptr,
                                                 unsigned tile_w, unsigned tile_h,
                                                 unsigned row_stride_elems)
{
  unsigned long long ga = (unsigned long long)gptr;
  u32x4 g0;
  g0[0] = 1u;                                         // count=1 (valid, user)
  g0[1] = lds_off_bytes;                              // lds_addr
  g0[2] = (unsigned)(ga & 0xffffffffu);               // global_addr[31:0]
  g0[3] = (unsigned)((ga >> 32) & 0x01ffffffu)        // global_addr[56:32]
        | 0x80000000u;                                // type=2 ("image")
  u32x8 g1;
  g1[0] = (1u << 16)        // data_size = 1 -> 2 bytes
        | (1u << 20)        // pad_enable
        | (5u << 22)        // pad_interval: 2^(5+1)=64 DWORDs (one 256B row)
        | (3u << 25);       // pad_amount: 3+1 = 4 DWORDs (8 halves)
  g1[1] = (tile_w & 0xffffu) << 16;  // tensor_dim0[15:0]
  g1[2] = (tile_h & 0xffffu) << 16;  // tensor_dim0[31:16]=0 | tensor_dim1[15:0]
  g1[3] = (tile_w & 0xffffu) << 16;  // tensor_dim1[31:16]=0 | tile_dim0
  g1[4] = (tile_h & 0xffffu);        // tile_dim1 | tile_dim2=0
  g1[5] = row_stride_elems;          // tensor_dim0_stride[31:0]
  g1[6] = 0u;
  g1[7] = 0u;
  asm volatile("tensor_load_to_lds %0, %1" :: "s"(g0), "s"(g1) : "memory");
}

// ---------- fp32 -> bf16 conversion ----------

__global__ void __launch_bounds__(256)
f32_to_bf16_kernel(const float* __restrict__ src, u16* __restrict__ dst, size_t n) {
  size_t i = (size_t)blockIdx.x * blockDim.x + threadIdx.x;
  size_t stride = (size_t)gridDim.x * blockDim.x;
  for (; i < n; i += stride) dst[i] = f2bf(src[i]);
}

// ---------- WMMA GEMM:  C[M,N] = A[M,K] * B[N,K]^T  (bf16 in, fp32 acc) ----------

enum { EPI_F32 = 0, EPI_BF16 = 1, EPI_ROPE = 2 };

template <int EPI>
__global__ void __launch_bounds__(256)
gemm_bf16_wmma(const u16* __restrict__ A, const u16* __restrict__ Bw,
               void* __restrict__ Cout, const long long* __restrict__ pos,
               int N, int K)
{
  const int lane   = threadIdx.x & 31;
  const int w      = threadIdx.x >> 5;
  const int laneLo = lane & 15;
  const int hi     = lane >> 4;
  const int wm     = w & 3;
  const int wn     = w >> 2;
  const int mbase  = blockIdx.x * 128 + wm * 32;
  const int nbase  = blockIdx.y * 256 + wn * 128;

  v8f acc[2][8];
#pragma unroll
  for (int i = 0; i < 2; ++i)
#pragma unroll
    for (int t = 0; t < 8; ++t) acc[i][t] = vzero8();

  const u16* arow0 = A + (size_t)(mbase      + laneLo) * K + (hi ? 8 : 0);
  const u16* arow1 = A + (size_t)(mbase + 16 + laneLo) * K + (hi ? 8 : 0);
  const u16* brow  = Bw + (size_t)(nbase + laneLo) * K + (hi ? 8 : 0);

  for (int k0 = 0; k0 < K; k0 += 32) {
    v16bf a0 = mkfrag(*(const v8u*)(arow0 + k0), *(const v8u*)(arow0 + k0 + 16));
    v16bf a1 = mkfrag(*(const v8u*)(arow1 + k0), *(const v8u*)(arow1 + k0 + 16));
#pragma unroll
    for (int t = 0; t < 8; ++t) {
      const u16* bp = brow + (size_t)(16 * t) * K + k0;
      v16bf b = mkfrag(*(const v8u*)bp, *(const v8u*)(bp + 16));
      acc[0][t] = __builtin_amdgcn_wmma_f32_16x16x32_bf16(false, a0, false, b, (short)0, acc[0][t], false, false);
      acc[1][t] = __builtin_amdgcn_wmma_f32_16x16x32_bf16(false, a1, false, b, (short)0, acc[1][t], false, false);
    }
  }

#pragma unroll
  for (int i = 0; i < 2; ++i) {
#pragma unroll
    for (int r = 0; r < 8; ++r) {
      int row = mbase + 16 * i + r + hi * 8;
      if (EPI == EPI_ROPE) {
        float p = (float)pos[row];
        u16* orow = (u16*)Cout + (size_t)row * N + nbase;
#pragma unroll
        for (int t = 0; t < 4; ++t) {
          int j = 16 * t + laneLo;
          float fr = p * __powf(10000.0f, -(float)j * (1.0f / 64.0f));
          float sn, cs;
          __sincosf(fr, &sn, &cs);
          float e0 = acc[i][t][r], e1 = acc[i][t + 4][r];
          orow[16 * t + laneLo]      = f2bf(e0 * cs - e1 * sn);
          orow[16 * t + 64 + laneLo] = f2bf(e1 * cs + e0 * sn);
        }
      } else if (EPI == EPI_BF16) {
        u16* orow = (u16*)Cout + (size_t)row * N + nbase;
#pragma unroll
        for (int t = 0; t < 8; ++t) orow[16 * t + laneLo] = f2bf(acc[i][t][r]);
      } else {
        float* orow = (float*)Cout + (size_t)row * N + nbase;
#pragma unroll
        for (int t = 0; t < 8; ++t) orow[16 * t + laneLo] = acc[i][t][r];
      }
    }
  }
}

// ---------- flash attention (GQA, causal) ----------
// LDS: Kt[2] double-buffered K tiles (TDM-loaded, padded rows),
//      Vt transposed V tile (manual staging), Pb per-wave P tiles.

__global__ void __launch_bounds__(256)
flash_attn(const u16* __restrict__ Q, const u16* __restrict__ Kc,
           const u16* __restrict__ Vc, u16* __restrict__ O)
{
  extern __shared__ u16 smem[];
  u16* Kt = smem;                        // 2 * TILE_H  (row-major [key][hd], padded)
  u16* Vt = smem + 2 * TILE_H;           // TILE_H      (transposed  [hd][key], padded)
  const int lane   = threadIdx.x & 31;
  const int w      = threadIdx.x >> 5;
  const int laneLo = lane & 15;
  const int hi     = lane >> 4;
  u16* Pb = smem + 3 * TILE_H + w * (16 * TROW);

  const int qb    = blockIdx.x;
  const int h     = blockIdx.y;
  const int kvh   = h >> 2;
  const int qrow0 = qb * 128 + w * 16;

  // Q fragments (head-dim 128 -> 4 K-steps of 32)
  v16bf qf[4];
  {
    const u16* qrow = Q + (size_t)(qrow0 + laneLo) * EQ + h * HDIM + (hi ? 8 : 0);
#pragma unroll
    for (int ks = 0; ks < 4; ++ks)
      qf[ks] = mkfrag(*(const v8u*)(qrow + ks * 32), *(const v8u*)(qrow + ks * 32 + 16));
  }

  v8f o[8];
  float m[8], l[8];
#pragma unroll
  for (int t = 0; t < 8; ++t) o[t] = vzero8();
#pragma unroll
  for (int r = 0; r < 8; ++r) { m[r] = -1e30f; l[r] = 0.0f; }

  const unsigned ktBase = (unsigned)(unsigned long long)(unsigned long long)(uintptr_t)Kt;

  // prologue: wave 0 kicks off TDM for K tile 0
  if (w == 0) {
    tdm_load_2d_bf16(ktBase, Kc + (size_t)0 * EKV + kvh * HDIM, 128, 128, EKV);
  }

  for (int kb = 0; kb <= qb; ++kb) {
    const u16* KtCur = Kt + (kb & 1) * TILE_H;

    // stage V tile transposed into LDS: Vt[col][key]
#pragma unroll
    for (int it = 0; it < 8; ++it) {
      int c    = threadIdx.x + it * 256;
      int krow = c >> 4;
      int ch   = c & 15;
      v8u d = *(const v8u*)(Vc + (size_t)(kb * 128 + krow) * EKV + kvh * HDIM + ch * 8);
#pragma unroll
      for (int i = 0; i < 8; ++i) Vt[(ch * 8 + i) * TROW + krow] = d[i];
    }
    // prefetch a slice of the next V tile while we compute this block
    if (kb < qb) {
      const u16* nv = Vc + (size_t)((kb + 1) * 128 + (threadIdx.x >> 1)) * EKV
                         + kvh * HDIM + (threadIdx.x & 1) * 64;
      __builtin_prefetch(nv, 0, 1);
    }
    if (w == 0) {
      __builtin_amdgcn_s_wait_tensorcnt(0);   // K tile kb resident
      if (kb < qb) {                          // overlap: DMA tile kb+1 into other buffer
        tdm_load_2d_bf16(ktBase + (unsigned)(((kb + 1) & 1) * TILE_H) * 2u,
                         Kc + (size_t)((kb + 1) * 128) * EKV + kvh * HDIM,
                         128, 128, EKV);
      }
    }
    __syncthreads();

    // S = Q * K^T (16x128): K fragments from TDM-staged LDS tile
    v8f sc[8];
#pragma unroll
    for (int t = 0; t < 8; ++t) sc[t] = vzero8();
#pragma unroll
    for (int t = 0; t < 8; ++t) {
      const u16* krow = KtCur + (16 * t + laneLo) * TROW + (hi ? 8 : 0);
#pragma unroll
      for (int ks = 0; ks < 4; ++ks) {
        v16bf b = mkfrag(*(const v8u*)(krow + ks * 32), *(const v8u*)(krow + ks * 32 + 16));
        sc[t] = __builtin_amdgcn_wmma_f32_16x16x32_bf16(false, qf[ks], false, b, (short)0, sc[t], false, false);
      }
    }

    // scale + causal mask + online softmax; P -> LDS (bf16, A-fragment layout)
    const float scale = 0.08838834764831845f;  // 1/sqrt(128)
#pragma unroll
    for (int r = 0; r < 8; ++r) {
      int rq = qrow0 + r + hi * 8;
      float mx = -1e30f;
#pragma unroll
      for (int t = 0; t < 8; ++t) {
        float v = sc[t][r] * scale;
        if (kb == qb) {
          int ck = kb * 128 + 16 * t + laneLo;
          if (ck > rq) v = -1e30f;
        }
        sc[t][r] = v;
        mx = fmaxf(mx, v);
      }
      mx = rmax16(mx);
      float nm    = fmaxf(m[r], mx);
      float alpha = __expf(m[r] - nm);
      m[r] = nm;
      float rs = 0.0f;
#pragma unroll
      for (int t = 0; t < 8; ++t) {
        float p = __expf(sc[t][r] - nm);
        sc[t][r] = p;
        rs += p;
      }
      rs   = rsum16(rs);
      l[r] = l[r] * alpha + rs;
#pragma unroll
      for (int t = 0; t < 8; ++t) o[t][r] *= alpha;
      int prow = r + hi * 8;
#pragma unroll
      for (int t = 0; t < 8; ++t) Pb[prow * TROW + 16 * t + laneLo] = f2bf(sc[t][r]);
    }

    // O += P * V
#pragma unroll
    for (int ks = 0; ks < 4; ++ks) {
      const u16* pr = Pb + laneLo * TROW + ks * 32 + (hi ? 8 : 0);
      v16bf a = mkfrag(*(const v8u*)pr, *(const v8u*)(pr + 16));
#pragma unroll
      for (int t = 0; t < 8; ++t) {
        const u16* vp = Vt + (16 * t + laneLo) * TROW + ks * 32 + (hi ? 8 : 0);
        v16bf b = mkfrag(*(const v8u*)vp, *(const v8u*)(vp + 16));
        o[t] = __builtin_amdgcn_wmma_f32_16x16x32_bf16(false, a, false, b, (short)0, o[t], false, false);
      }
    }
    __syncthreads();
  }

  // normalize and store attention output (bf16, (S, H*HD) row-major)
#pragma unroll
  for (int r = 0; r < 8; ++r) {
    int row = qrow0 + r + hi * 8;
    float inv = 1.0f / l[r];
    u16* orow = O + (size_t)row * EQ + h * HDIM;
#pragma unroll
    for (int t = 0; t < 8; ++t) orow[16 * t + laneLo] = f2bf(o[t][r] * inv);
  }
}

// ---------- launch ----------

extern "C" void kernel_launch(void* const* d_in, const int* in_sizes, int n_in,
                              void* d_out, int out_size, void* d_ws, size_t ws_size,
                              hipStream_t stream)
{
  const float*     hs  = (const float*)d_in[0];
  const long long* pos = (const long long*)d_in[1];
  const float*     Wq  = (const float*)d_in[2];
  const float*     Wk  = (const float*)d_in[3];
  const float*     Wv  = (const float*)d_in[4];
  const float*     Wo  = (const float*)d_in[5];

  u16* p   = (u16*)d_ws;
  u16* Xb  = p; p += (size_t)S_LEN * D_MODEL;
  u16* Wqb = p; p += (size_t)EQ * D_MODEL;
  u16* Wkb = p; p += (size_t)EKV * D_MODEL;
  u16* Wvb = p; p += (size_t)EKV * D_MODEL;
  u16* Wob = p; p += (size_t)D_MODEL * EQ;
  u16* Qb  = p; p += (size_t)S_LEN * EQ;
  u16* Kb  = p; p += (size_t)S_LEN * EKV;
  u16* Vb  = p; p += (size_t)S_LEN * EKV;
  u16* Ab  = p; p += (size_t)S_LEN * EQ;

  dim3 blk(256);
  f32_to_bf16_kernel<<<2048, blk, 0, stream>>>(hs, Xb, (size_t)S_LEN * D_MODEL);
  f32_to_bf16_kernel<<<4096, blk, 0, stream>>>(Wq, Wqb, (size_t)EQ * D_MODEL);
  f32_to_bf16_kernel<<<1024, blk, 0, stream>>>(Wk, Wkb, (size_t)EKV * D_MODEL);
  f32_to_bf16_kernel<<<1024, blk, 0, stream>>>(Wv, Wvb, (size_t)EKV * D_MODEL);
  f32_to_bf16_kernel<<<4096, blk, 0, stream>>>(Wo, Wob, (size_t)D_MODEL * EQ);

  gemm_bf16_wmma<EPI_ROPE><<<dim3(S_LEN / 128, EQ  / 256), blk, 0, stream>>>(Xb, Wqb, Qb, pos, EQ,  D_MODEL);
  gemm_bf16_wmma<EPI_ROPE><<<dim3(S_LEN / 128, EKV / 256), blk, 0, stream>>>(Xb, Wkb, Kb, pos, EKV, D_MODEL);
  gemm_bf16_wmma<EPI_BF16><<<dim3(S_LEN / 128, EKV / 256), blk, 0, stream>>>(Xb, Wvb, Vb, nullptr, EKV, D_MODEL);

  // LDS: 2 K tiles (TDM) + V tile + 8 wave P tiles, all 136-half padded rows
  size_t lds = (size_t)(3 * TILE_H + 8 * 16 * TROW) * sizeof(u16);  // ~139 KB
  flash_attn<<<dim3(S_LEN / 128, NH), blk, lds, stream>>>(Qb, Kb, Vb, Ab);

  gemm_bf16_wmma<EPI_F32><<<dim3(S_LEN / 128, D_MODEL / 256), blk, 0, stream>>>(Ab, Wob, d_out, nullptr, D_MODEL, D_MODEL);
}
